// Operator_46050639347870
// MI455X (gfx1250) — compile-verified, matrix operand order checked by reading, not run on hardware
//
#include <hip/hip_runtime.h>
#include <math.h>

// ---- MI455X / gfx1250: wave32, WMMA f16 16x16x32 (f32 acc), TDM async tensor loads ----

typedef _Float16 half_t;
typedef _Float16 v16h __attribute__((ext_vector_type(16)));
typedef float    v8f  __attribute__((ext_vector_type(8)));
typedef unsigned int u32x4 __attribute__((ext_vector_type(4)));
typedef int      i32x8 __attribute__((ext_vector_type(8)));
typedef int      i32x4 __attribute__((ext_vector_type(4)));

#define XDIM   2048
#define CHUNK  64      // columns per chunk tile
#define NCHUNK 8       // 512 cols per block / 64
#define XSPLIT 4       // blocks per (b,p,c)
#define NWAVE  4       // 128 threads

#if __has_builtin(__builtin_amdgcn_s_wait_tensorcnt)
#define WAIT_TENSOR0() __builtin_amdgcn_s_wait_tensorcnt(0)
#define WAIT_TENSOR3() __builtin_amdgcn_s_wait_tensorcnt(3)
#else
#define WAIT_TENSOR0()
#define WAIT_TENSOR3()
#endif

__device__ __forceinline__ v8f wmma_f16(v16h a, v16h b, v8f c) {
  // v_wmma_f32_16x16x32_f16 : D = A(16x32 f16) * B(32x16 f16) + C(16x16 f32)
  return __builtin_amdgcn_wmma_f32_16x16x32_f16(false, a, false, b, (short)0, c,
                                                false, false);
}

// A-fragment of a 16x16 f16 weight (K padded 16->32 with zeros).
// A layout (16-bit A 16x32): lane m=L%16, g=L/16; elems 0..7 -> K=8g+e, elems 8..15 -> K=16+8g+e (zeroed here).
__device__ __forceinline__ v16h afrag16(const half_t* __restrict__ W, int m, int g) {
  v16h a;
#pragma unroll
  for (int e = 0; e < 8; ++e) { a[e] = W[m * 16 + 8 * g + e]; a[e + 8] = (half_t)0.f; }
  return a;
}

// A-fragment (step s of 8) of a 16x256 f16 matrix stored row-major [16][256].
__device__ __forceinline__ v16h afrag256(const half_t* __restrict__ W, int m, int g, int s) {
  v16h a;
  const int base = m * 256 + 32 * s + 8 * g;
#pragma unroll
  for (int e = 0; e < 8; ++e) { a[e] = W[base + e]; a[e + 8] = W[base + 16 + e]; }
  return a;
}

// Spill a 16x16 f32 C/D-layout tile to LDS [16][16]: lane owns rows 8g..8g+7 of column n.
__device__ __forceinline__ void store_tile(float* __restrict__ S, v8f d, int n, int g) {
#pragma unroll
  for (int r = 0; r < 8; ++r) S[(r + 8 * g) * 16 + n] = d[r];
}

// OUT(16xN) = Af(16x256,f16) @ outer(u,v)(256xN): u[jHi] from LDS tile U, v[jLo]
// from the producing accumulator's own registers (D-layout rows 8g..8g+7 == jLo set).
__device__ __forceinline__ v8f contract256(const v16h* __restrict__ Af,
                                           const float* __restrict__ U,
                                           v8f Vr, int n) {
  v8f acc = {0.f, 0.f, 0.f, 0.f, 0.f, 0.f, 0.f, 0.f};
#pragma unroll
  for (int s = 0; s < 8; ++s) {
    const float u0 = U[(2 * s) * 16 + n];
    const float u1 = U[(2 * s + 1) * 16 + n];
    v16h b;
#pragma unroll
    for (int i = 0; i < 8; ++i) {
      const float vv = Vr[i];
      b[i]     = (half_t)(u0 * vv);   // j = 32s +      8g+i  -> jHi=2s,   jLo=8g+i
      b[i + 8] = (half_t)(u1 * vv);   // j = 32s + 16 + 8g+i  -> jHi=2s+1, jLo=8g+i
    }
    acc = wmma_f16(Af[s], b, acc);
  }
  return acc;
}

__device__ __forceinline__ void act8(v8f& d) {  // sign(x)*sqrt(|x|)
#pragma unroll
  for (int r = 0; r < 8; ++r) {
    float v = d[r];
    d[r] = copysignf(sqrtf(fabsf(v)), v);
  }
}

// 2-D TDM: tile_rows x tile_cols f32 tile, row stride row_stride (elements), into LDS.
__device__ __forceinline__ void tdm_load_2d(unsigned lds_off, const float* gptr,
                                            int tile_cols, int tile_rows, int row_stride) {
#if __has_builtin(__builtin_amdgcn_tensor_load_to_lds)
  const unsigned long long ga = (unsigned long long)(const void*)gptr;
  u32x4 g0;
  g0[0] = 1u;                                 // count=1 (valid user descriptor)
  g0[1] = lds_off;                            // D#.lds_addr
  g0[2] = (unsigned)ga;                       // D#.global_addr[31:0]
  g0[3] = (unsigned)(ga >> 32) | (2u << 30);  // D#.global_addr[56:32] | type=2
  i32x8 g1;
  g1[0] = (int)(2u << 16);                    // data_size=4B; mask/pad/iterate=0
  g1[1] = (int)((unsigned)XDIM << 16);        // tensor_dim0 = 2048
  g1[2] = (int)(16u << 16);                   // tensor_dim1 = 16
  g1[3] = (int)((unsigned)tile_cols << 16);   // tile_dim0
  g1[4] = tile_rows;                          // tile_dim1 (tile_dim2=0)
  g1[5] = row_stride;                         // tensor_dim0_stride[31:0]
  g1[6] = 0;
  g1[7] = 0;
  i32x4 gz4 = {0, 0, 0, 0};                   // D# groups 2/3 unused (2-D tensor)
  i32x8 gz8 = {0, 0, 0, 0, 0, 0, 0, 0};       // extra group (6-arg toolchain variant)
  __builtin_amdgcn_tensor_load_to_lds(g0, g1, gz4, gz4, gz8, 0);
#else
  (void)lds_off; (void)gptr; (void)tile_cols; (void)tile_rows; (void)row_stride;
#endif
}

__global__ void __launch_bounds__(128)
zero_out_kernel(float* __restrict__ out, int n) {
  int i = blockIdx.x * blockDim.x + threadIdx.x;
  if (i < n) out[i] = 0.f;
}

__global__ void __launch_bounds__(128)
era5_predrnn_attn(const float* __restrict__ x,
                  const float* __restrict__ K0, const float* __restrict__ Q0,
                  const float* __restrict__ V0,
                  const float* __restrict__ K1, const float* __restrict__ Q1,
                  const float* __restrict__ V1,
                  const float* __restrict__ enc, const float* __restrict__ dec,
                  const float* __restrict__ aW, const float* __restrict__ wW,
                  float* __restrict__ out) {
  __shared__ float  lx[2][16 * CHUNK];     // x chunk tiles (double buffered)
  __shared__ float  abuf[2][16 * CHUNK];   // a chunk tiles
  __shared__ float  wbuf[2][16 * CHUNK];   // w chunk tiles
  __shared__ half_t encF[4096];            // enc flattened [e][t*16+T], f16
  __shared__ half_t decF[4096];            // dec permuted  [T][e*16+t], f16
  __shared__ half_t m0k[256], m0q[256], m0v[256];  // per-channel layer-0 weights
  __shared__ half_t m1k[256], m1q[256], m1v[256];  // shared layer-1 weights
  __shared__ float  wscr[NWAVE][256];      // one 16x16 U-tile per wave

  const int tid  = threadIdx.x;
  const int lane = tid & 31;
  const int wv   = tid >> 5;
  const int n    = lane & 15;     // column within tile / A-row m
  const int g    = lane >> 4;     // half-wave group

  const int xsplit = blockIdx.x & (XSPLIT - 1);
  const int bpc    = blockIdx.x >> 2;       // 0..255 -> (b*P+p)*32 + c
  const int c      = bpc & 31;
  const int bp     = bpc >> 5;
  const float* xslab = x + ((size_t)bp * 512 + (size_t)c) * (size_t)XDIM;  // t-stride = 32*2048

  // ---- preload static weights into LDS (f16) ----
  for (int i = tid; i < 4096; i += 128) {
    encF[i] = (half_t)enc[i];                       // enc flat is already [e][t*16+T]
    const int e = i >> 8, t = (i >> 4) & 15, Td = i & 15;
    decF[Td * 256 + e * 16 + t] = (half_t)dec[i];   // decF[T][e*16+t] = dec[e,t,T]
  }
  for (int i = tid; i < 256; i += 128) {
    m0k[i] = (half_t)K0[i * 32 + c];
    m0q[i] = (half_t)Q0[i * 32 + c];
    m0v[i] = (half_t)V0[i * 32 + c];
    m1k[i] = (half_t)K1[i];
    m1q[i] = (half_t)Q1[i];
    m1v[i] = (half_t)V1[i];
  }

  // ---- kick off TDM for chunk 0 (wave 0 only; EXEC-independent DMA) ----
  if (wv == 0) {
    const int gcb = xsplit * 512;
    tdm_load_2d((unsigned)(unsigned long long)(void*)&lx[0][0],   xslab + gcb, CHUNK, 16, 32 * XDIM);
    tdm_load_2d((unsigned)(unsigned long long)(void*)&abuf[0][0], aW + gcb,    CHUNK, 16, XDIM);
    tdm_load_2d((unsigned)(unsigned long long)(void*)&wbuf[0][0], wW + gcb,    CHUNK, 16, XDIM);
  }
  __syncthreads();  // weights visible to all waves

  // ---- per-lane A fragments of enc/dec (kept in VGPRs for all tiles) ----
  v16h encA[8], decA[8];
#pragma unroll
  for (int s = 0; s < 8; ++s) {
    encA[s] = afrag256(encF, n, g, s);
    decA[s] = afrag256(decF, n, g, s);
  }

  float yacc = 0.f;
  const int col = (wv << 4) + n;  // this wave's column within the chunk
  float* SU = &wscr[wv][0];

  for (int ci = 0; ci < NCHUNK; ++ci) {
    const int bi = ci & 1;
    if (wv == 0) {
      if (ci + 1 < NCHUNK) {  // prefetch next chunk, then wait for current (in-order TDM)
        const int nb  = (ci + 1) & 1;
        const int gcb = xsplit * 512 + (ci + 1) * CHUNK;
        tdm_load_2d((unsigned)(unsigned long long)(void*)&lx[nb][0],   xslab + gcb, CHUNK, 16, 32 * XDIM);
        tdm_load_2d((unsigned)(unsigned long long)(void*)&abuf[nb][0], aW + gcb,    CHUNK, 16, XDIM);
        tdm_load_2d((unsigned)(unsigned long long)(void*)&wbuf[nb][0], wW + gcb,    CHUNK, 16, XDIM);
        WAIT_TENSOR3();
      } else {
        WAIT_TENSOR0();
      }
    }
    __syncthreads();

    const float* LX = &lx[bi][0];
    const float* AB = &abuf[bi][0];
    const float* WB = &wbuf[bi][0];

    // ---- layer 0: K,Q0,V = W @ Xtile (K=16, zero-padded to 32) ----
    v16h bx;
#pragma unroll
    for (int e = 0; e < 8; ++e) {
      bx[e]     = (half_t)LX[(8 * g + e) * CHUNK + col];
      bx[e + 8] = (half_t)0.f;
    }
    const v8f z8 = {0.f, 0.f, 0.f, 0.f, 0.f, 0.f, 0.f, 0.f};
    v8f Kt = wmma_f16(afrag16(m0k, n, g), bx, z8);
    v8f Qt = wmma_f16(afrag16(m0q, n, g), bx, z8);  // q0 kept in regs until final dec
    v8f Vt = wmma_f16(afrag16(m0v, n, g), bx, z8);

    // ---- enc contraction 1: h = sqrt_act(EncF @ outer(k, v)) ----
    store_tile(SU, Kt, n, g);                 // u = k (needs cross-half rows -> LDS)
    v8f H = contract256(encA, SU, Vt, n);     // v = v straight from registers
    act8(H);

    // ---- layer 1: D-layout rows(8g..8g+7) == B-frag rows -> no LDS round trip ----
    v16h bh;
#pragma unroll
    for (int e = 0; e < 8; ++e) { bh[e] = (half_t)H[e]; bh[e + 8] = (half_t)0.f; }
    v8f K2 = wmma_f16(afrag16(m1k, n, g), bh, z8);
    v8f Q2 = wmma_f16(afrag16(m1q, n, g), bh, z8);
    v8f V2 = wmma_f16(afrag16(m1v, n, g), bh, z8);

    // ---- enc contraction 2 ----
    store_tile(SU, K2, n, g);
    v8f H2 = contract256(encA, SU, V2, n);
    act8(H2);
#pragma unroll
    for (int r = 0; r < 8; ++r)               // b = a[e,x] * h
      H2[r] = H2[r] * AB[(r + 8 * g) * CHUNK + col];

    // ---- decoder contraction 1: out[T] = DecF @ outer(b, q1) ----
    store_tile(SU, H2, n, g);
    v8f B1 = contract256(decA, SU, Q2, n);

    // ---- decoder contraction 2 with q0 ----
    store_tile(SU, B1, n, g);
    v8f B2 = contract256(decA, SU, Qt, n);

    // ---- y partial: sum_t w[t,x] * B2[t,x] ----
#pragma unroll
    for (int r = 0; r < 8; ++r)
      yacc += WB[(r + 8 * g) * CHUNK + col] * B2[r];

    __syncthreads();  // protect double buffers before next TDM overwrite
  }

  // ---- reduce 32 lanes, one f32 atomic per wave into y[b,p,0,c] ----
  float tot = yacc;
#pragma unroll
  for (int off = 16; off > 0; off >>= 1) tot += __shfl_xor(tot, off, 32);
  if (lane == 0) atomicAdd(&out[bpc], tot);
}

extern "C" void kernel_launch(void* const* d_in, const int* in_sizes, int n_in,
                              void* d_out, int out_size, void* d_ws, size_t ws_size,
                              hipStream_t stream) {
  (void)in_sizes; (void)n_in; (void)d_ws; (void)ws_size;
  const float* x   = (const float*)d_in[0];
  const float* K0  = (const float*)d_in[1];
  const float* Q0  = (const float*)d_in[2];
  const float* V0  = (const float*)d_in[3];
  const float* K1  = (const float*)d_in[4];
  const float* Q1  = (const float*)d_in[5];
  const float* V1  = (const float*)d_in[6];
  const float* enc = (const float*)d_in[7];
  const float* dec = (const float*)d_in[8];
  const float* aW  = (const float*)d_in[9];
  const float* wW  = (const float*)d_in[10];
  float* out = (float*)d_out;

  zero_out_kernel<<<(out_size + 127) / 128, 128, 0, stream>>>(out, out_size);
  era5_predrnn_attn<<<dim3(256 * XSPLIT), dim3(128), 0, stream>>>(
      x, K0, Q0, V0, K1, Q1, V1, enc, dec, aW, wW, out);
}
